// Rnn_12086037971649
// MI455X (gfx1250) — compile-verified, matrix operand order checked by reading, not run on hardware
//
#include <hip/hip_runtime.h>
#include <hip/hip_bf16.h>

// Problem constants (match reference)
#define BB   128
#define SSQ  512
#define DIN  128
#define HH   512
#define CC   1000
#define CPAD 1008   // 63 * 16

typedef __attribute__((ext_vector_type(16))) __bf16 v16bf;
typedef __attribute__((ext_vector_type(8)))  __bf16 v8bf;
typedef __attribute__((ext_vector_type(8)))  float  v8f;

// Build a 16x32 bf16 A/B fragment for one lane.
// Lane l (l<16): row/col = l, K = {kbase..kbase+7, kbase+16..kbase+23}
// Lane l (l>=16): row/col = l-16, K shifted by +8 (khalf folded into ptr by caller)
static __device__ __forceinline__ v16bf make_frag(const __bf16* p) {
  v8bf lo = *(const v8bf*)(p);
  v8bf hi = *(const v8bf*)(p + 16);
  v16bf r;
#pragma unroll
  for (int i = 0; i < 8; ++i) { r[i] = lo[i]; r[i + 8] = hi[i]; }
  return r;
}

// ---------------- init: zero h0 and grid-barrier counters ----------------
__global__ void k_init(__bf16* __restrict__ h0, unsigned* __restrict__ cnt) {
  int i = blockIdx.x * blockDim.x + threadIdx.x;
  if (i < BB * HH) h0[i] = (__bf16)0.0f;
  if (i < 64) cnt[i] = 0u;
}

// ---------------- fp32 -> bf16 conversion ----------------
__global__ void k_cvt(const float* __restrict__ src, __bf16* __restrict__ dst, long n) {
  long i = (long)blockIdx.x * blockDim.x + threadIdx.x;
  long stride = (long)gridDim.x * blockDim.x;
  for (; i < n; i += stride) dst[i] = (__bf16)src[i];
}

// fp32 -> bf16 with row padding (zero rows beyond `rows`)
__global__ void k_cvt_pad(const float* __restrict__ src, __bf16* __restrict__ dst,
                          int rows, int cols, int prows) {
  long n = (long)prows * cols;
  long i = (long)blockIdx.x * blockDim.x + threadIdx.x;
  long stride = (long)gridDim.x * blockDim.x;
  for (; i < n; i += stride) {
    long rr = i / cols;
    dst[i] = (rr < rows) ? (__bf16)src[i] : (__bf16)0.0f;
  }
}

// ---------------- xp = A[MxK] @ W[NxK]^T + b1 + b2  (fp32 out) ----------------
// Register-blocked 64x16 strip per wave (4 M-tiles x 1 N-tile). K-loop is
// unrolled x2 with SSA buffers (no cross-iteration rotation -> no reg copies):
// each iteration issues 10 independent b128-pair fragment loads, then 8 WMMAs.
// K is always a multiple of 64 here. 8 waves per block.
__global__ void __launch_bounds__(256)
k_gemm_bias(const __bf16* __restrict__ A, const __bf16* __restrict__ W,
            const float* __restrict__ b1, const float* __restrict__ b2,
            float* __restrict__ out, int M, int N, int K) {
  const int lane = threadIdx.x & 31;
  const int wave = threadIdx.x >> 5;
  const int ntn = N >> 4;                         // N tiles
  const long job = (long)blockIdx.x * 8 + wave;   // one 64x16 strip per wave
  const long total = (long)(M >> 6) * ntn;
  if (job >= total) return;
  const int mg = (int)(job / ntn);                // M group (64 rows)
  const int tn = (int)(job % ntn);
  const int r = lane & 15;
  const int khalf = (lane >> 4) * 8;

  const __bf16* arow0 = A + (size_t)(mg * 64 + r) * K + khalf;  // +j*16*K per M-tile
  const __bf16* wrow  = W + (size_t)(tn * 16 + r) * K + khalf;

  v8f acc[4] = {};

  for (int ks = 0; ks < K; ks += 64) {
    // two K-steps of fragments, all fresh SSA values (10 load pairs in flight)
    v16bf b0 = make_frag(wrow + ks);
    v16bf a0[4];
#pragma unroll
    for (int j = 0; j < 4; ++j) a0[j] = make_frag(arow0 + (size_t)j * 16 * K + ks);
    v16bf b1v = make_frag(wrow + ks + 32);
    v16bf a1[4];
#pragma unroll
    for (int j = 0; j < 4; ++j) a1[j] = make_frag(arow0 + (size_t)j * 16 * K + ks + 32);

#pragma unroll
    for (int j = 0; j < 4; ++j)
      acc[j] = __builtin_amdgcn_wmma_f32_16x16x32_bf16(false, a0[j], false, b0,
                                                       (short)0, acc[j], false, false);
#pragma unroll
    for (int j = 0; j < 4; ++j)
      acc[j] = __builtin_amdgcn_wmma_f32_16x16x32_bf16(false, a1[j], false, b1v,
                                                       (short)0, acc[j], false, false);
  }

  const int col = tn * 16 + r;
  const float bias = b1[col] + b2[col];
#pragma unroll
  for (int j = 0; j < 4; ++j) {
    const int rbase = mg * 64 + j * 16 + ((lane >> 4) << 3);
#pragma unroll
    for (int i = 0; i < 8; ++i)
      out[(size_t)(rbase + i) * N + col] = acc[j][i] + bias;
  }
}

// ---------------- persistent recurrent layer ----------------
// h_t = tanh(xp[:,t,:] + h_{t-1} @ Whh^T), stored bf16 into hseq[b][t][:].
// Grid: 32 blocks x 256 threads = 256 waves = 256 tiles (8 Mtiles x 32 Ntiles).
// Whh fragments live in 128 VGPRs for all 512 steps; each step bursts all 16
// A fragments into another 128 VGPRs, then fires 16 back-to-back WMMAs.
__global__ void __launch_bounds__(256)
k_rnn_layer(const __bf16* __restrict__ Whh, const float* __restrict__ xp,
            const __bf16* __restrict__ h0, __bf16* __restrict__ hseq,
            unsigned* __restrict__ cnt) {
  const int lane = threadIdx.x & 31;
  const int wave = threadIdx.x >> 5;
  const int id = blockIdx.x * 8 + wave;   // 0..255
  const int tm = id >> 5;                 // 0..7   (batch tile)
  const int tn = id & 31;                 // 0..31  (hidden tile)
  const int r = lane & 15;
  const int khalf = (lane >> 4) * 8;

  // Preload B operand (Whh rows are contiguous along K): 16 K-steps.
  v16bf bw[16];
  const __bf16* wrow = Whh + (size_t)(tn * 16 + r) * HH + khalf;
#pragma unroll
  for (int ks = 0; ks < 16; ++ks) bw[ks] = make_frag(wrow + ks * 32);

  const int abrow = tm * 16 + r;                 // batch row this lane loads
  const int col = tn * 16 + r;                   // output column
  const int rbase = tm * 16 + ((lane >> 4) << 3);

  for (int t = 0; t < SSQ; ++t) {
    const __bf16* arow = (t == 0)
        ? (h0 + (size_t)abrow * HH + khalf)
        : (hseq + ((size_t)abrow * SSQ + (t - 1)) * HH + khalf);

    // burst-load the whole 16x512 A strip for this lane (32 b128 loads)
    v16bf af[16];
#pragma unroll
    for (int ks = 0; ks < 16; ++ks) af[ks] = make_frag(arow + ks * 32);

    v8f acc = {};
#pragma unroll
    for (int ks = 0; ks < 16; ++ks)
      acc = __builtin_amdgcn_wmma_f32_16x16x32_bf16(false, af[ks], false, bw[ks],
                                                    (short)0, acc, false, false);

#pragma unroll
    for (int i = 0; i < 8; ++i) {
      size_t idx = ((size_t)(rbase + i) * SSQ + t) * HH + col;
      // xp is consumed exactly once: non-temporal load keeps L2 for Whh/hseq
      float v = acc[i] + __builtin_nontemporal_load(&xp[idx]);
      hseq[idx] = (__bf16)tanhf(v);
    }
    // device-scope grid barrier (monotonic counter; 32 co-resident WGs)
    __threadfence();
    __syncthreads();
    if (threadIdx.x == 0) {
      __hip_atomic_fetch_add(cnt, 1u, __ATOMIC_RELEASE, __HIP_MEMORY_SCOPE_AGENT);
      const unsigned target = 32u * (unsigned)(t + 1);
      while (__hip_atomic_load(cnt, __ATOMIC_ACQUIRE, __HIP_MEMORY_SCOPE_AGENT) < target)
        __builtin_amdgcn_s_sleep(1);
    }
    __syncthreads();
    __threadfence();
  }
}

// ---------------- logits = h_last @ Wout^T + b_out ----------------
// 63 blocks x 8 waves = 504 tiles (8 Mtiles x 63 Ntiles over padded C).
__global__ void __launch_bounds__(256)
k_gemm_logits(const __bf16* __restrict__ hseq, const __bf16* __restrict__ Wout,
              const float* __restrict__ bout, float* __restrict__ logits) {
  const int lane = threadIdx.x & 31;
  const int wave = threadIdx.x >> 5;
  const int tile = blockIdx.x * 8 + wave;  // 0..503
  const int tm = tile & 7;
  const int tn = tile >> 3;                // 0..62
  const int r = lane & 15;
  const int khalf = (lane >> 4) * 8;

  const __bf16* arow = hseq + ((size_t)(tm * 16 + r) * SSQ + (SSQ - 1)) * HH + khalf;
  const __bf16* wrow = Wout + (size_t)(tn * 16 + r) * HH + khalf;

  v8f acc = {};
#pragma unroll
  for (int ks = 0; ks < HH; ks += 32) {   // fully unrolled: SSA, no rotation
    v16bf a = make_frag(arow + ks);
    v16bf b = make_frag(wrow + ks);
    acc = __builtin_amdgcn_wmma_f32_16x16x32_bf16(false, a, false, b,
                                                  (short)0, acc, false, false);
  }

  const int col = tn * 16 + r;
  if (col < CC) {
    const float bias = bout[col];
    const int rbase = tm * 16 + ((lane >> 4) << 3);
#pragma unroll
    for (int i = 0; i < 8; ++i)
      logits[(size_t)(rbase + i) * CC + col] = acc[i] + bias;
  }
}

// ---------------- row-wise log_softmax over C=1000 ----------------
__global__ void __launch_bounds__(256)
k_logsoftmax(const float* __restrict__ logits, float* __restrict__ out) {
  __shared__ float red[256];
  const int row = blockIdx.x;
  const float* x = logits + (size_t)row * CC;
  float m = -3.4e38f;
  for (int c = threadIdx.x; c < CC; c += 256) m = fmaxf(m, x[c]);
  red[threadIdx.x] = m; __syncthreads();
  for (int s = 128; s > 0; s >>= 1) {
    if (threadIdx.x < s) red[threadIdx.x] = fmaxf(red[threadIdx.x], red[threadIdx.x + s]);
    __syncthreads();
  }
  const float mx = red[0]; __syncthreads();
  float sum = 0.f;
  for (int c = threadIdx.x; c < CC; c += 256) sum += expf(x[c] - mx);
  red[threadIdx.x] = sum; __syncthreads();
  for (int s = 128; s > 0; s >>= 1) {
    if (threadIdx.x < s) red[threadIdx.x] += red[threadIdx.x + s];
    __syncthreads();
  }
  const float lse = mx + logf(red[0]);
  for (int c = threadIdx.x; c < CC; c += 256)
    out[(size_t)row * CC + c] = x[c] - lse;
}

extern "C" void kernel_launch(void* const* d_in, const int* in_sizes, int n_in,
                              void* d_out, int out_size, void* d_ws, size_t ws_size,
                              hipStream_t stream) {
  (void)in_sizes; (void)n_in; (void)out_size; (void)ws_size;
  const float* x     = (const float*)d_in[0];
  const float* W_ih0 = (const float*)d_in[1];
  const float* W_hh0 = (const float*)d_in[2];
  const float* b_ih0 = (const float*)d_in[3];
  const float* b_hh0 = (const float*)d_in[4];
  const float* W_ih1 = (const float*)d_in[5];
  const float* W_hh1 = (const float*)d_in[6];
  const float* b_ih1 = (const float*)d_in[7];
  const float* b_hh1 = (const float*)d_in[8];
  const float* W_out = (const float*)d_in[9];
  const float* b_out = (const float*)d_in[10];

  // Workspace carve-out (~287 MB)
  char* ws = (char*)d_ws;
  size_t off = 0;
  auto carve = [&](size_t bytes) -> void* {
    void* p = ws + off;
    off += (bytes + 255) & ~(size_t)255;
    return p;
  };
  float*  xp     = (float*) carve((size_t)BB * SSQ * HH * 4);   // 134 MB (reused both layers)
  __bf16* xbf    = (__bf16*)carve((size_t)BB * SSQ * DIN * 2);  // 16.8 MB
  __bf16* hseq0  = (__bf16*)carve((size_t)BB * SSQ * HH * 2);   // 67 MB
  __bf16* hseq1  = (__bf16*)carve((size_t)BB * SSQ * HH * 2);   // 67 MB
  __bf16* wih0b  = (__bf16*)carve((size_t)HH * DIN * 2);
  __bf16* whh0b  = (__bf16*)carve((size_t)HH * HH * 2);
  __bf16* wih1b  = (__bf16*)carve((size_t)HH * HH * 2);
  __bf16* whh1b  = (__bf16*)carve((size_t)HH * HH * 2);
  __bf16* woutb  = (__bf16*)carve((size_t)CPAD * HH * 2);
  __bf16* h0     = (__bf16*)carve((size_t)BB * HH * 2);
  float*  logits = (float*) carve((size_t)BB * CC * 4);
  unsigned* cnt  = (unsigned*)carve(256);

  const dim3 blk(256);

  k_init<<<(BB * HH + 255) / 256, blk, 0, stream>>>(h0, cnt);
  k_cvt<<<2048, blk, 0, stream>>>(x, xbf, (long)BB * SSQ * DIN);
  k_cvt<<<64,   blk, 0, stream>>>(W_ih0, wih0b, (long)HH * DIN);
  k_cvt<<<256,  blk, 0, stream>>>(W_hh0, whh0b, (long)HH * HH);
  k_cvt<<<256,  blk, 0, stream>>>(W_ih1, wih1b, (long)HH * HH);
  k_cvt<<<256,  blk, 0, stream>>>(W_hh1, whh1b, (long)HH * HH);
  k_cvt_pad<<<512, blk, 0, stream>>>(W_out, woutb, CC, HH, CPAD);

  const int M = BB * SSQ;                                   // 65536
  const int xp_blocks = ((M / 64) * (HH / 16)) / 8;         // 4096 (64x16 strip per wave)

  // Layer 0
  k_gemm_bias<<<xp_blocks, blk, 0, stream>>>(xbf, wih0b, b_ih0, b_hh0, xp, M, HH, DIN);
  k_rnn_layer<<<32, blk, 0, stream>>>(whh0b, xp, h0, hseq0, cnt + 0);

  // Layer 1
  k_gemm_bias<<<xp_blocks, blk, 0, stream>>>(hseq0, wih1b, b_ih1, b_hh1, xp, M, HH, HH);
  k_rnn_layer<<<32, blk, 0, stream>>>(whh1b, xp, h0, hseq1, cnt + 16);

  // Classifier + log_softmax
  k_gemm_logits<<<63, blk, 0, stream>>>(hseq1, woutb, b_out, logits);
  k_logsoftmax<<<BB, blk, 0, stream>>>(logits, (float*)d_out);
}
